// BlockMoEAdapters_5506148073586
// MI455X (gfx1250) — compile-verified, matrix operand order checked by reading, not run on hardware
//
#include <hip/hip_runtime.h>
#include <hip/hip_bf16.h>
#include <math.h>
#include <stdint.h>

// ---------------------------------------------------------------------------
// Types for CDNA5 WMMA (wave32): D(f32 16x16) = A(bf16 16x32) x B(bf16 32x16) + C
// ---------------------------------------------------------------------------
typedef __attribute__((ext_vector_type(16))) __bf16 v16bf;
typedef __attribute__((ext_vector_type(8)))  float  v8f;

static __device__ __forceinline__ v8f wmma_bf16(v16bf a, v16bf b, v8f c) {
  // (neg_a, A, neg_b, B, c_mod, C, reuse_a, reuse_b)
  return __builtin_amdgcn_wmma_f32_16x16x32_bf16(false, a, false, b, (short)0, c,
                                                 false, false);
}

// CDNA5 async copy: global -> LDS, 16 bytes per lane, tracked by ASYNCcnt.
// lds_off = low 32 bits of the generic pointer (ISA: LDS generic addresses
// carry the LDS byte offset in [31:0]).
static __device__ __forceinline__ void async_ld_b128(void* lds_ptr,
                                                     const void* gptr) {
  const unsigned lds_off = (unsigned)(uintptr_t)lds_ptr;
  asm volatile("global_load_async_to_lds_b128 %0, %1, off"
               :
               : "v"(lds_off), "v"(gptr)
               : "memory");
}
static __device__ __forceinline__ void wait_async0() {
  asm volatile("s_wait_asynccnt 0" ::: "memory");
}

// Load a 16x32 bf16 fragment (A layout; B "Bt" layout is identical with N<->M)
// from row-major LDS with leading dim `ld` (bf16 elements, multiple of 8).
// Lane l: row = base + (l&15); half = l>>4 selects K subranges
// [half*8, half*8+8) and [16+half*8, 16+half*8+8)  -> two aligned b128 loads.
static __device__ __forceinline__ v16bf frag16(const __bf16* base, int row,
                                               int kbase, int ld) {
  const int lane = threadIdx.x & 31;
  const __bf16* p =
      base + (size_t)(row + (lane & 15)) * ld + kbase + ((lane >> 4) * 8);
  union { uint4 u[2]; v16bf v; } t;
  t.u[0] = *(const uint4*)p;
  t.u[1] = *(const uint4*)(p + 16);
  return t.v;
}

static __device__ __forceinline__ float gelu_exact(float x) {
  return 0.5f * x * (1.0f + erff(x * 0.70710678118654752f));
}

// ---------------------------------------------------------------------------
// Generic bf16 WMMA GEMM:  C[M][N] = A[M][K] * Bt[N][K]^T  (+bias, gelu, resid)
// Block 256 threads = 8 waves; tile 128(M) x 64(N); K-step 32.
// Async double-buffered LDS: global_load_async_to_lds_b128 fills the next
// tile while WMMA consumes the current one; one barrier per K-step.
// Epilogue modes: f32 store, bf16 store, atomic scatter-add with row scale.
// Requires M%128==0, N%64==0, K%32==0 (true for every GEMM in this model).
// ---------------------------------------------------------------------------
__global__ __launch_bounds__(256) void gemm_bf16_kernel(
    const __bf16* __restrict__ A, const __bf16* __restrict__ Bt,
    const float* __restrict__ bias, const float* __restrict__ residual,
    float* Cf, __bf16* Cb, const int* __restrict__ scat_rows,
    const float* __restrict__ scat_scale, float* scat_out, int M, int N, int K,
    int act) {
  __shared__ __align__(16) __bf16 As[2][128][48];  // K32 tile + pad to 96B rows
  __shared__ __align__(16) __bf16 Bs[2][64][48];

  const int tid = threadIdx.x;
  const int wave = tid >> 5, lane = tid & 31;
  const int wm = wave & 3, wn = wave >> 2;
  const int bm = blockIdx.y * 128, bn = blockIdx.x * 64;

  const int ra = tid >> 1, ca = (tid & 1) * 16;   // A: 16 bf16 per thread
  const int rb = tid >> 2, cb = (tid & 3) * 8;    // B: 8 bf16 per thread

  auto issue_tile = [&](int buf, int k0) {
    const __bf16* ga = A + (size_t)(bm + ra) * K + k0 + ca;
    async_ld_b128(&As[buf][ra][ca], ga);
    async_ld_b128(&As[buf][ra][ca + 8], ga + 8);
    const __bf16* gb = Bt + (size_t)(bn + rb) * K + k0 + cb;
    async_ld_b128(&Bs[buf][rb][cb], gb);
  };

  v8f acc[2][2] = {};
  issue_tile(0, 0);
  int cur = 0;

  for (int k0 = 0; k0 < K; k0 += 32) {
    wait_async0();      // current tile landed in LDS (this wave's copies)
    __syncthreads();    // all waves done copying AND done reading other buffer
    if (k0 + 32 < K) issue_tile(cur ^ 1, k0 + 32);  // overlap with compute

    const v16bf a0 = frag16(&As[cur][0][0], wm * 32, 0, 48);
    const v16bf a1 = frag16(&As[cur][0][0], wm * 32 + 16, 0, 48);
    const v16bf b0 = frag16(&Bs[cur][0][0], wn * 32, 0, 48);
    const v16bf b1 = frag16(&Bs[cur][0][0], wn * 32 + 16, 0, 48);
    acc[0][0] = wmma_bf16(a0, b0, acc[0][0]);
    acc[0][1] = wmma_bf16(a0, b1, acc[0][1]);
    acc[1][0] = wmma_bf16(a1, b0, acc[1][0]);
    acc[1][1] = wmma_bf16(a1, b1, acc[1][1]);
    cur ^= 1;
  }

  // Epilogue. C layout: lane half (l>>4) + VGPR v -> row half*8+v; col = l&15.
  const int half = lane >> 4, c0 = lane & 15;
#pragma unroll
  for (int i = 0; i < 2; ++i)
#pragma unroll
    for (int j = 0; j < 2; ++j)
#pragma unroll
      for (int v = 0; v < 8; ++v) {
        const int row = bm + wm * 32 + i * 16 + half * 8 + v;
        const int col = bn + wn * 32 + j * 16 + c0;
        float val = acc[i][j][v];
        if (bias) val += bias[col];
        if (act) val = gelu_exact(val);
        if (residual) val += residual[(size_t)row * N + col];
        if (Cf) Cf[(size_t)row * N + col] = val;
        if (Cb) Cb[(size_t)row * N + col] = (__bf16)val;
        if (scat_out) {
          const float sc = scat_scale[row];
          atomicAdd(&scat_out[(size_t)scat_rows[row] * N + col], val * sc);
        }
      }
}

// ---------------------------------------------------------------------------
// Flash attention: grid (N/64, H, B), 128 threads = 4 waves.
// Wave w owns 16 query rows x full 64-col key chunk -> row stats stay in-wave.
// Q/K tiles staged with async-to-LDS; V transposed manually (scattered b16).
// qkv layout: [T][3][H][hd] bf16;  out: [T][H*hd] bf16.
// ---------------------------------------------------------------------------
__global__ __launch_bounds__(128) void attn_kernel(
    const __bf16* __restrict__ qkv, __bf16* __restrict__ out, int N, int Hh,
    int hd) {
  const int D3 = 3 * Hh * hd;   // 3072
  const int Dm = Hh * hd;       // 1024
  const int qb = blockIdx.x, h = blockIdx.y, b = blockIdx.z;
  const int tid = threadIdx.x, wave = tid >> 5, lane = tid & 31;
  const int half = lane >> 4, c0 = lane & 15;

  __shared__ __align__(16) __bf16 Qs[64][80];
  __shared__ __align__(16) __bf16 Ks[64][80];
  __shared__ __align__(16) __bf16 Vt[64][80];       // V transposed: [d][key]
  __shared__ __align__(16) __bf16 Pb[4][16][80];    // per-wave P staging

  const int r = tid >> 1, d0 = (tid & 1) * 32;
  {  // Q tile (64x64) via async copy; first in-loop wait covers it
    const __bf16* g = qkv + (size_t)(b * N + qb * 64 + r) * D3 + h * hd + d0;
#pragma unroll
    for (int u = 0; u < 4; ++u) async_ld_b128(&Qs[r][d0 + u * 8], g + u * 8);
  }

  v8f o[4] = {};
  float mstat[8], lstat[8];
#pragma unroll
  for (int v = 0; v < 8; ++v) { mstat[v] = -1e30f; lstat[v] = 0.f; }

  for (int kb = 0; kb < N / 64; ++kb) {
    __syncthreads();  // prior P@V reads of Ks/Vt complete before overwrite
    {
      const __bf16* gk =
          qkv + (size_t)(b * N + kb * 64 + r) * D3 + Dm + h * hd + d0;
#pragma unroll
      for (int u = 0; u < 4; ++u)
        async_ld_b128(&Ks[r][d0 + u * 8], gk + u * 8);
      const __bf16* gv =
          qkv + (size_t)(b * N + kb * 64 + r) * D3 + 2 * Dm + h * hd + d0;
#pragma unroll
      for (int u = 0; u < 32; ++u) Vt[d0 + u][r] = gv[u];  // transpose to LDS
    }
    wait_async0();
    __syncthreads();

    // S = (Q * hd^-0.5) @ K^T  (64x64 per block; 16x64 per wave)
    v8f s[4] = {};
#pragma unroll
    for (int ks = 0; ks < 64; ks += 32) {
      const v16bf a = frag16(&Qs[0][0], wave * 16, ks, 80);
#pragma unroll
      for (int j = 0; j < 4; ++j) {
        const v16bf bf = frag16(&Ks[0][0], j * 16, ks, 80);
        s[j] = wmma_bf16(a, bf, s[j]);
      }
    }

    // online softmax per row (row = half*8+v, fully within this wave)
#pragma unroll
    for (int v = 0; v < 8; ++v) {
      float rmax = -1e30f;
#pragma unroll
      for (int j = 0; j < 4; ++j) {
        s[j][v] *= 0.125f;  // hd^-0.5
        rmax = fmaxf(rmax, s[j][v]);
      }
#pragma unroll
      for (int off = 1; off < 16; off <<= 1)
        rmax = fmaxf(rmax, __shfl_xor(rmax, off, 16));
      const float mnew = fmaxf(mstat[v], rmax);
      const float alpha = __expf(mstat[v] - mnew);
      float rsum = 0.f;
#pragma unroll
      for (int j = 0; j < 4; ++j) {
        const float p = __expf(s[j][v] - mnew);
        s[j][v] = p;
        rsum += p;
      }
#pragma unroll
      for (int off = 1; off < 16; off <<= 1) rsum += __shfl_xor(rsum, off, 16);
      lstat[v] = lstat[v] * alpha + rsum;
      mstat[v] = mnew;
#pragma unroll
      for (int j = 0; j < 4; ++j) {
        o[j][v] *= alpha;
        Pb[wave][half * 8 + v][j * 16 + c0] = (__bf16)s[j][v];
      }
    }
    __syncthreads();  // Pb visible for re-fragmentation

    // O += P @ V
#pragma unroll
    for (int ks = 0; ks < 64; ks += 32) {
      const v16bf a = frag16(&Pb[wave][0][0], 0, ks, 80);
#pragma unroll
      for (int j = 0; j < 4; ++j) {
        const v16bf bf = frag16(&Vt[0][0], j * 16, ks, 80);
        o[j] = wmma_bf16(a, bf, o[j]);
      }
    }
  }

  const int trow = b * N + qb * 64 + wave * 16 + half * 8;
#pragma unroll
  for (int v = 0; v < 8; ++v) {
    const float inv = 1.0f / lstat[v];
#pragma unroll
    for (int j = 0; j < 4; ++j)
      out[(size_t)(trow + v) * Dm + h * hd + j * 16 + c0] =
          (__bf16)(o[j][v] * inv);
  }
}

// ---------------------------------------------------------------------------
// LayerNorm (one token per block). Optional f32 and/or bf16 outputs.
// ---------------------------------------------------------------------------
__global__ __launch_bounds__(256) void ln_kernel(const float* __restrict__ x,
                                                 const float* __restrict__ g,
                                                 const float* __restrict__ bta,
                                                 float* outf, __bf16* outb,
                                                 int D) {
  const int t = blockIdx.x;
  const float* row = x + (size_t)t * D;
  float s = 0.f, ss = 0.f;
  for (int d = threadIdx.x; d < D; d += 256) {
    const float v = row[d];
    s += v;
    ss += v * v;
  }
#pragma unroll
  for (int off = 16; off; off >>= 1) {
    s += __shfl_xor(s, off, 32);
    ss += __shfl_xor(ss, off, 32);
  }
  __shared__ float rs[8], rss[8], mv[2];
  const int wave = threadIdx.x >> 5, lane = threadIdx.x & 31;
  if (lane == 0) { rs[wave] = s; rss[wave] = ss; }
  __syncthreads();
  if (threadIdx.x == 0) {
    float a = 0.f, b2 = 0.f;
    for (int i = 0; i < 8; ++i) { a += rs[i]; b2 += rss[i]; }
    const float mu = a / D;
    mv[0] = mu;
    mv[1] = rsqrtf(b2 / D - mu * mu + 1e-5f);
  }
  __syncthreads();
  const float mu = mv[0], r = mv[1];
  for (int d = threadIdx.x; d < D; d += 256) {
    const float v = (row[d] - mu) * r * g[d] + bta[d];
    if (outf) outf[(size_t)t * D + d] = v;
    if (outb) outb[(size_t)t * D + d] = (__bf16)v;
  }
}

// ---------------------------------------------------------------------------
// Router: noisy top-2 over E=4 experts. One wave per token.
// ---------------------------------------------------------------------------
__global__ __launch_bounds__(128) void router_kernel(
    const float* __restrict__ x2, const float* __restrict__ noise,
    const float* __restrict__ w_route, const float* __restrict__ b_route,
    const float* __restrict__ w_noise, const float* __restrict__ b_noise,
    float* gate, int* fidx, int D) {
  const int wave = threadIdx.x >> 5, lane = threadIdx.x & 31;
  const int t = blockIdx.x * 4 + wave;
  float ar[4] = {0, 0, 0, 0}, an[4] = {0, 0, 0, 0};
  for (int d = lane; d < D; d += 32) {
    const float xv = x2[(size_t)t * D + d];
#pragma unroll
    for (int e = 0; e < 4; ++e) {
      ar[e] += xv * w_route[d * 4 + e];
      an[e] += xv * w_noise[d * 4 + e];
    }
  }
#pragma unroll
  for (int e = 0; e < 4; ++e)
#pragma unroll
    for (int off = 16; off; off >>= 1) {
      ar[e] += __shfl_xor(ar[e], off, 32);
      an[e] += __shfl_xor(an[e], off, 32);
    }
  if (lane == 0) {
    float noisy[4];
#pragma unroll
    for (int e = 0; e < 4; ++e) {
      const float lg = ar[e] + b_route[e];
      const float ns = an[e] + b_noise[e];
      const float sp = (ns > 20.f) ? ns : log1pf(__expf(ns));
      noisy[e] = lg + noise[t * 4 + e] * sp;
    }
    int i0 = 0;
    for (int e = 1; e < 4; ++e)
      if (noisy[e] > noisy[i0]) i0 = e;
    int i1 = -1;
    for (int e = 0; e < 4; ++e)
      if (e != i0 && (i1 < 0 || noisy[e] > noisy[i1])) i1 = e;
    const float m = fmaxf(noisy[i0], noisy[i1]);
    const float e0 = __expf(noisy[i0] - m), e1 = __expf(noisy[i1] - m);
    const float p0 = e0 / (e0 + e1);
#pragma unroll
    for (int e = 0; e < 4; ++e) gate[t * 4 + e] = 0.f;
    gate[t * 4 + i0] = p0;
    gate[t * 4 + i1] = 1.f - p0;
    fidx[2 * t] = i0;
    fidx[2 * t + 1] = i1;
  }
}

// ---------------------------------------------------------------------------
// Per-expert capacity dispatch (stable token order). grid = E blocks.
// ---------------------------------------------------------------------------
__global__ __launch_bounds__(256) void dispatch_scan(
    const int* __restrict__ fidx, const float* __restrict__ gate, int* sel,
    float* selw, int capacity, int T) {
  const int e = blockIdx.x, tid = threadIdx.x;
  __shared__ int sm[256];
  int base = 0;
  for (int c0 = 0; c0 < T; c0 += 256) {
    const int t = c0 + tid;
    int m = 0;
    if (t < T) m = (fidx[2 * t] == e || fidx[2 * t + 1] == e) ? 1 : 0;
    sm[tid] = m;
    __syncthreads();
    for (int off = 1; off < 256; off <<= 1) {  // inclusive Hillis-Steele
      const int x = sm[tid];
      const int y = (tid >= off) ? sm[tid - off] : 0;
      __syncthreads();
      sm[tid] = x + y;
      __syncthreads();
    }
    const int incl = sm[tid];
    const int tot = sm[255];
    __syncthreads();
    const int pos = base + incl - 1;
    if (m && pos < capacity) {
      sel[e * capacity + pos] = t;
      selw[e * capacity + pos] = gate[t * 4 + e];
    }
    base += tot;
  }
  for (int i = (base < capacity ? base : capacity) + tid; i < capacity;
       i += 256) {
    sel[e * capacity + i] = 0;
    selw[e * capacity + i] = 0.f;  // padded rows contribute nothing
  }
}

__global__ __launch_bounds__(256) void gather_kernel(
    const __bf16* __restrict__ x2b, const int* __restrict__ sel, __bf16* xdisp,
    int D) {
  const int row = blockIdx.x;
  const int src = sel[row];
  for (int d = threadIdx.x; d < D; d += 256)
    xdisp[(size_t)row * D + d] = x2b[(size_t)src * D + d];
}

// ---------------------------------------------------------------------------
// fp32 -> bf16 converters (direct and transposing, for [K][N] -> [N][K]).
// ---------------------------------------------------------------------------
__global__ __launch_bounds__(256) void cvt_bf16(const float* __restrict__ in,
                                                __bf16* out, size_t n) {
  size_t i = (size_t)blockIdx.x * 256 + threadIdx.x;
  const size_t stride = (size_t)gridDim.x * 256;
  for (; i < n; i += stride) out[i] = (__bf16)in[i];
}

__global__ __launch_bounds__(256) void cvt_bf16_t(const float* __restrict__ in,
                                                  __bf16* out, int K, int N) {
  const size_t n = (size_t)K * N;
  size_t i = (size_t)blockIdx.x * 256 + threadIdx.x;
  const size_t stride = (size_t)gridDim.x * 256;
  for (; i < n; i += stride) {
    const int k = (int)(i / N), c = (int)(i % N);
    out[(size_t)c * K + k] = (__bf16)in[i];
  }
}

// ---------------------------------------------------------------------------
// Launch orchestration.
// ---------------------------------------------------------------------------
extern "C" void kernel_launch(void* const* d_in, const int* in_sizes, int n_in,
                              void* d_out, int out_size, void* d_ws,
                              size_t ws_size, hipStream_t stream) {
  (void)in_sizes; (void)n_in; (void)out_size; (void)ws_size;
  constexpr int Bb = 2, Nn = 2048, Dd = 1024, Hh = 16, hd = 64;
  constexpr int T = Bb * Nn, D3 = 3 * Dd, MLPH = 4096, MOEH = 256;
  constexpr int Ee = 4, CAP = 2048;

  const float* x      = (const float*)d_in[0];
  const float* noise  = (const float*)d_in[1];
  const float* ln1_g  = (const float*)d_in[2];
  const float* ln1_b  = (const float*)d_in[3];
  const float* ln2_g  = (const float*)d_in[4];
  const float* ln2_b  = (const float*)d_in[5];
  const float* w_qkv  = (const float*)d_in[6];
  const float* w_proj = (const float*)d_in[7];
  const float* b_proj = (const float*)d_in[8];
  const float* w_rt   = (const float*)d_in[9];
  const float* b_rt   = (const float*)d_in[10];
  const float* w_nz   = (const float*)d_in[11];
  const float* b_nz   = (const float*)d_in[12];
  const float* we1    = (const float*)d_in[13];
  const float* be1    = (const float*)d_in[14];
  const float* we2    = (const float*)d_in[15];
  const float* be2    = (const float*)d_in[16];
  const float* w_m1   = (const float*)d_in[17];
  const float* b_m1   = (const float*)d_in[18];
  const float* w_m2   = (const float*)d_in[19];
  const float* b_m2   = (const float*)d_in[20];
  float* out = (float*)d_out;

  char* wsp = (char*)d_ws;
  size_t off = 0;
  auto alloc = [&](size_t bytes) {
    void* p = wsp + off;
    off = (off + bytes + 255) & ~(size_t)255;
    return p;
  };
  __bf16* x1b    = (__bf16*)alloc((size_t)T * Dd * 2);
  __bf16* wqkvb  = (__bf16*)alloc((size_t)D3 * Dd * 2);
  __bf16* qkvb   = (__bf16*)alloc((size_t)T * D3 * 2);
  __bf16* attnb  = (__bf16*)alloc((size_t)T * Dd * 2);
  __bf16* wprojb = (__bf16*)alloc((size_t)Dd * Dd * 2);
  float*  xattn  = (float*)alloc((size_t)T * Dd * 4);
  float*  x2f    = (float*)alloc((size_t)T * Dd * 4);
  __bf16* x2b    = (__bf16*)alloc((size_t)T * Dd * 2);
  __bf16* wm1t   = (__bf16*)alloc((size_t)Dd * MLPH * 2);
  __bf16* wm2t   = (__bf16*)alloc((size_t)MLPH * Dd * 2);
  __bf16* hmlp   = (__bf16*)alloc((size_t)T * MLPH * 2);
  __bf16* we1t   = (__bf16*)alloc((size_t)Ee * Dd * MOEH * 2);
  __bf16* we2t   = (__bf16*)alloc((size_t)Ee * MOEH * Dd * 2);
  __bf16* xdisp  = (__bf16*)alloc((size_t)Ee * CAP * Dd * 2);
  __bf16* hdisp  = (__bf16*)alloc((size_t)Ee * CAP * MOEH * 2);
  float*  gate   = (float*)alloc((size_t)T * Ee * 4);
  int*    fidx   = (int*)alloc((size_t)T * 2 * 4);
  int*    sel    = (int*)alloc((size_t)Ee * CAP * 4);
  float*  selw   = (float*)alloc((size_t)Ee * CAP * 4);

  // 1) weight conversion to bf16 (w_qkv/w_proj are already [N][K])
  cvt_bf16<<<2048, 256, 0, stream>>>(w_qkv, wqkvb, (size_t)D3 * Dd);
  cvt_bf16<<<1024, 256, 0, stream>>>(w_proj, wprojb, (size_t)Dd * Dd);
  cvt_bf16_t<<<2048, 256, 0, stream>>>(w_m1, wm1t, Dd, MLPH);
  cvt_bf16_t<<<2048, 256, 0, stream>>>(w_m2, wm2t, MLPH, Dd);
  for (int e = 0; e < Ee; ++e) {
    cvt_bf16_t<<<512, 256, 0, stream>>>(we1 + (size_t)e * Dd * MOEH,
                                        we1t + (size_t)e * MOEH * Dd, Dd, MOEH);
    cvt_bf16_t<<<512, 256, 0, stream>>>(we2 + (size_t)e * MOEH * Dd,
                                        we2t + (size_t)e * Dd * MOEH, MOEH, Dd);
  }

  // 2) LN1 -> x1 (bf16)
  ln_kernel<<<T, 256, 0, stream>>>(x, ln1_g, ln1_b, nullptr, x1b, Dd);

  // 3) QKV GEMM: [T,D] x [3D,D]^T -> bf16 [T,3D]
  gemm_bf16_kernel<<<dim3(D3 / 64, T / 128), 256, 0, stream>>>(
      x1b, wqkvb, nullptr, nullptr, nullptr, qkvb, nullptr, nullptr, nullptr,
      T, D3, Dd, 0);

  // 4) flash attention -> bf16 [T,D]
  attn_kernel<<<dim3(Nn / 64, Hh, Bb), 128, 0, stream>>>(qkvb, attnb, Nn, Hh,
                                                         hd);

  // 5) proj GEMM + b_proj + residual x -> xattn (f32)
  gemm_bf16_kernel<<<dim3(Dd / 64, T / 128), 256, 0, stream>>>(
      attnb, wprojb, b_proj, x, xattn, nullptr, nullptr, nullptr, nullptr, T,
      Dd, Dd, 0);

  // 6) LN2 -> x2 (f32 + bf16)
  ln_kernel<<<T, 256, 0, stream>>>(xattn, ln2_g, ln2_b, x2f, x2b, Dd);

  // 7) router + capacity dispatch + gather
  router_kernel<<<T / 4, 128, 0, stream>>>(x2f, noise, w_rt, b_rt, w_nz, b_nz,
                                           gate, fidx, Dd);
  dispatch_scan<<<Ee, 256, 0, stream>>>(fidx, gate, sel, selw, CAP, T);
  gather_kernel<<<Ee * CAP, 256, 0, stream>>>(x2b, sel, xdisp, Dd);

  // 8) dense MLP: gelu(x2@W1+b1)@W2+b2 + xattn -> d_out (f32)
  gemm_bf16_kernel<<<dim3(MLPH / 64, T / 128), 256, 0, stream>>>(
      x2b, wm1t, b_m1, nullptr, nullptr, hmlp, nullptr, nullptr, nullptr, T,
      MLPH, Dd, 1);
  gemm_bf16_kernel<<<dim3(Dd / 64, T / 128), 256, 0, stream>>>(
      hmlp, wm2t, b_m2, xattn, out, nullptr, nullptr, nullptr, nullptr, T, Dd,
      MLPH, 0);

  // 9) expert FFNs; second GEMM scatter-adds gate-weighted rows into d_out
  for (int e = 0; e < Ee; ++e) {
    gemm_bf16_kernel<<<dim3(MOEH / 64, CAP / 128), 256, 0, stream>>>(
        xdisp + (size_t)e * CAP * Dd, we1t + (size_t)e * MOEH * Dd,
        be1 + (size_t)e * MOEH, nullptr, nullptr,
        hdisp + (size_t)e * CAP * MOEH, nullptr, nullptr, nullptr, CAP, MOEH,
        Dd, 1);
    gemm_bf16_kernel<<<dim3(Dd / 64, CAP / 128), 256, 0, stream>>>(
        hdisp + (size_t)e * CAP * MOEH, we2t + (size_t)e * Dd * MOEH,
        be2 + (size_t)e * Dd, nullptr, nullptr, nullptr, sel + e * CAP,
        selw + e * CAP, out, CAP, Dd, MOEH, 0);
  }
}
